// FalconAttentionFused_8804682957417
// MI455X (gfx1250) — matrix operand order repaired and checked by reading, not compiled, not run on hardware
//
#include <hip/hip_runtime.h>
#include <stdint.h>

#define DEV __device__ __forceinline__

typedef __attribute__((ext_vector_type(16))) __bf16 v16bf;
typedef __attribute__((ext_vector_type(8)))  float  v8f;
typedef unsigned int  u32x4 __attribute__((ext_vector_type(4)));
typedef int           i32x4 __attribute__((ext_vector_type(4)));
typedef int           i32x8 __attribute__((ext_vector_type(8)));
typedef unsigned short u16;

// Problem sizes (B=1)
#define S_LEN 2048
#define HID   4544
#define NHEAD 71
#define HDIM  64
#define QKV_O 4672   // (71+2)*64
#define KSTEP 64

#if defined(__has_builtin)
#if __has_builtin(__builtin_amdgcn_tensor_load_to_lds) && __has_builtin(__builtin_amdgcn_s_wait_tensorcnt)
#define HAVE_TDM 1
#endif
#endif
#ifndef HAVE_TDM
#define HAVE_TDM 0
#endif

#if HAVE_TDM
#define WAIT_TENSORCNT(n) __builtin_amdgcn_s_wait_tensorcnt((short)(n))
#endif

union FragBF { uint4 q[2]; v16bf v; };
union U4U16  { uint4 q; u16 s[8]; };

DEV u16 f2bf(float f) {            // f32 -> bf16 round-to-nearest-even
  union { float f; unsigned u; } v; v.f = f;
  unsigned r = v.u + 0x7FFFu + ((v.u >> 16) & 1u);
  return (u16)(r >> 16);
}

DEV v16bf ld_frag(const u16* p0, const u16* p1) {   // two 16B chunks -> 16 bf16
  FragBF fr;
  fr.q[0] = *(const uint4*)p0;
  fr.q[1] = *(const uint4*)p1;
  return fr.v;
}

DEV v8f wmma_bf16(v16bf a, v16bf b, v8f c) {
  return __builtin_amdgcn_wmma_f32_16x16x32_bf16(false, a, false, b, (short)0, c,
                                                 false, false);
}

#if HAVE_TDM
// ---------------------------------------------------------------------------
// TDM 2D tile load, global -> LDS.  Tile: `rows` lines of `tile_d0_8B` 8-byte
// units, line stride `stride_8B` 8-byte units.  D# layout per CDNA5 ISA §8.
// Issue once per workgroup (wave 0); completion via TENSORcnt.
// ---------------------------------------------------------------------------
DEV void tdm_load_2d(unsigned lds_addr, const void* gptr,
                     unsigned tile_d0_8B, unsigned rows, unsigned stride_8B) {
  unsigned long long ga = (unsigned long long)gptr;
  u32x4 g0 = { 1u,                                   // count=1 valid descriptor
               lds_addr,                             // lds_addr [63:32]
               (unsigned)ga,                         // global_addr lo
               (unsigned)(ga >> 32) | 0x80000000u }; // global_addr hi | type=2
  i32x8 g1;
  g1[0] = 3 << 16;                                   // data_size=8B, wg_mask=0
  g1[1] = (int)((tile_d0_8B & 0xFFFFu) << 16);       // tensor_dim0[15:0]
  g1[2] = (int)((tile_d0_8B >> 16) | ((rows & 0xFFFFu) << 16)); // dim0 hi, dim1 lo
  g1[3] = (int)((rows >> 16) | (tile_d0_8B << 16));  // dim1 hi, tile_dim0
  g1[4] = (int)rows;                                 // tile_dim1 (tile_dim2=0)
  g1[5] = (int)stride_8B;                            // tensor_dim0_stride lo32
  g1[6] = 0;
  g1[7] = 0;
  i32x4 z = { 0, 0, 0, 0 };
#if __clang_major__ >= 23
  i32x8 z8 = { 0, 0, 0, 0, 0, 0, 0, 0 };
  __builtin_amdgcn_tensor_load_to_lds(g0, g1, z, z, z8, 0);
#else
  __builtin_amdgcn_tensor_load_to_lds(g0, g1, z, z, 0);
#endif
}
#endif

// ---------------------------------------------------------------------------
// f32 -> bf16 bulk convert (vectorized x4)
// ---------------------------------------------------------------------------
__global__ __launch_bounds__(256) void cvt_bf16_kernel(
    const float4* __restrict__ in, ushort4* __restrict__ out, int n4) {
  int i = blockIdx.x * 256 + threadIdx.x;
  if (i >= n4) return;
  float4 f = in[i];
  ushort4 o;
  o.x = f2bf(f.x); o.y = f2bf(f.y); o.z = f2bf(f.z); o.w = f2bf(f.w);
  out[i] = o;
}

// ---------------------------------------------------------------------------
// GEMM: C[M,N] = A[M,K] * W[N,K]^T, bf16 inputs, f32 accumulate.
// Block tile 128x64, K-step 64, 8 waves (4Mx2N), 32x32 per wave, 8 WMMA/stage.
// TDM path: double-buffered LDS, wave 0 issues DMA for tile kt+1 while all
// waves compute tile kt; pipelined with s_wait_tensorcnt<=2.
// ---------------------------------------------------------------------------
__global__ __launch_bounds__(256) void gemm_kernel(
    const u16* __restrict__ A, const u16* __restrict__ W,
    float* __restrict__ C, int M, int N, int K)
{
  __shared__ __align__(16) u16 As[2][128 * KSTEP];   // 2 x 16 KB
  __shared__ __align__(16) u16 Bs[2][64 * KSTEP];    // 2 x  8 KB

  const int tid   = threadIdx.x;
  const int m0    = blockIdx.y * 128;
  const int n0    = blockIdx.x * 64;
  const int wave  = tid >> 5, lane = tid & 31;
  const int wr    = wave & 3, wc = wave >> 2;
  const int lhalf = lane >> 4, lmod = lane & 15;

  v8f acc[2][2] = {};
  const int kTiles = K / KSTEP;

#if HAVE_TDM
  const unsigned strideA8 = (unsigned)(K >> 2);       // K*2 bytes / 8
  const unsigned aAs[2] = { (unsigned)(size_t)(void*)&As[0][0],
                            (unsigned)(size_t)(void*)&As[1][0] };
  const unsigned aBs[2] = { (unsigned)(size_t)(void*)&Bs[0][0],
                            (unsigned)(size_t)(void*)&Bs[1][0] };
  if (wave == 0) {   // prologue: DMA tile 0
    tdm_load_2d(aAs[0], A + (long)m0 * K, KSTEP / 4, 128, strideA8);
    tdm_load_2d(aBs[0], W + (long)n0 * K, KSTEP / 4, 64, strideA8);
  }
#endif

  for (int kt = 0; kt < kTiles; ++kt) {
    const int kb  = kt * KSTEP;
    const int cur = kt & 1;
#if HAVE_TDM
    if (wave == 0) {
      if (kt + 1 < kTiles) {   // DMA next tile into the other buffer
        tdm_load_2d(aAs[cur ^ 1], A + (long)m0 * K + kb + KSTEP, KSTEP / 4, 128, strideA8);
        tdm_load_2d(aBs[cur ^ 1], W + (long)n0 * K + kb + KSTEP, KSTEP / 4, 64, strideA8);
        WAIT_TENSORCNT(2);     // in-order: current tile's 2 DMAs complete
      } else {
        WAIT_TENSORCNT(0);
      }
    }
    asm volatile("" ::: "memory");
#else
    // manual staging fallback: A 128x64 (4 chunks/thread), B 64x64 (2/thread)
    #pragma unroll
    for (int i = 0; i < 4; ++i) {
      int ci = tid + i * 256, row = ci >> 3, cc = ci & 7;
      *(uint4*)&As[cur][row * KSTEP + cc * 8] =
          *(const uint4*)(A + (long)(m0 + row) * K + kb + cc * 8);
    }
    #pragma unroll
    for (int i = 0; i < 2; ++i) {
      int ci = tid + i * 256, row = ci >> 3, cc = ci & 7;
      *(uint4*)&Bs[cur][row * KSTEP + cc * 8] =
          *(const uint4*)(W + (long)(n0 + row) * K + kb + cc * 8);
    }
#endif
    __syncthreads();

    const u16* Ab = &As[cur][0];
    const u16* Bb = &Bs[cur][0];
    #pragma unroll
    for (int kc = 0; kc < 2; ++kc) {
      v16bf aF[2], bF[2];
      const int ac0 = kc * 32 + (lhalf ? 8 : 0);
      #pragma unroll
      for (int mf = 0; mf < 2; ++mf) {
        int row = wr * 32 + mf * 16 + lmod;
        aF[mf] = ld_frag(&Ab[row * KSTEP + ac0], &Ab[row * KSTEP + ac0 + 16]);
      }
      const int bk0 = kc * 32 + (lhalf ? 8 : 0);
      #pragma unroll
      for (int nf = 0; nf < 2; ++nf) {
        int nrow = wc * 32 + nf * 16 + lmod;
        bF[nf] = ld_frag(&Bb[nrow * KSTEP + bk0], &Bb[nrow * KSTEP + bk0 + 16]);
      }
      #pragma unroll
      for (int mf = 0; mf < 2; ++mf)
        #pragma unroll
        for (int nf = 0; nf < 2; ++nf)
          acc[mf][nf] = wmma_bf16(aF[mf], bF[nf], acc[mf][nf]);
    }
    __syncthreads();
  }

  // C layout: lane holds col n=lmod, rows r + 8*lhalf
  #pragma unroll
  for (int mf = 0; mf < 2; ++mf)
    #pragma unroll
    for (int nf = 0; nf < 2; ++nf)
      #pragma unroll
      for (int r = 0; r < 8; ++r) {
        int row = m0 + wr * 32 + mf * 16 + r + 8 * lhalf;
        int col = n0 + wc * 32 + nf * 16 + lmod;
        C[(long)row * N + col] = acc[mf][nf][r];
      }
}

// ---------------------------------------------------------------------------
// RoPE + head split: xqkv f32 [S,73,64] -> Qb bf16 [71][S][64] (rotated),
// Kb bf16 [S][64] (rotated), Vb bf16 [S][64].
// ---------------------------------------------------------------------------
__global__ __launch_bounds__(256) void rope_kernel(
    const float* __restrict__ xqkv, u16* __restrict__ Qb,
    u16* __restrict__ Kb, u16* __restrict__ Vb)
{
  int t  = blockIdx.x * 256 + threadIdx.x;  // S*73*32 threads
  int j  = t & 31;
  int r  = t >> 5;
  int hh = r % 73;
  int s  = r / 73;
  if (s >= S_LEN) return;
  long base = ((long)s * 73 + hh) * 64;
  float v0 = xqkv[base + j];
  float v1 = xqkv[base + j + 32];
  if (hh == 72) {                       // V head: passthrough
    Vb[s * 64 + j]      = f2bf(v0);
    Vb[s * 64 + j + 32] = f2bf(v1);
    return;
  }
  float invf = __expf((float)j * -0.2878231366242557f);  // 10000^(-j/32)
  float ang  = (float)s * invf;
  float sn, c;
  __sincosf(ang, &sn, &c);
  float o0 = v0 * c - v1 * sn;
  float o1 = v1 * c + v0 * sn;
  if (hh < NHEAD) {
    long ob = ((long)hh * S_LEN + s) * 64;
    Qb[ob + j]      = f2bf(o0);
    Qb[ob + j + 32] = f2bf(o1);
  } else {
    Kb[s * 64 + j]      = f2bf(o0);
    Kb[s * 64 + j + 32] = f2bf(o1);
  }
}

// ---------------------------------------------------------------------------
// Flash attention (causal, multi-query): grid (S/128, NH). 8 waves; each wave
// owns 16 query rows. K tiles DMA'd by TDM; V staged transposed by all waves.
// Online softmax with cross-lane reductions; P re-laid out C->A via LDS.
// ---------------------------------------------------------------------------
__global__ __launch_bounds__(256) void flash_kernel(
    const u16* __restrict__ Qb, const u16* __restrict__ Kb,
    const u16* __restrict__ Vb, u16* __restrict__ Ob)
{
  __shared__ __align__(16) u16 Ks[32 * 64];
  __shared__ __align__(16) u16 Vt[64 * 32];
  __shared__ __align__(16) u16 Ps[8][16 * 32];

  const int tid   = threadIdx.x;
  const int wave  = tid >> 5, lane = tid & 31;
  const int lhalf = lane >> 4, lmod = lane & 15;
  const int qblk  = blockIdx.x, h = blockIdx.y;
  const int q0    = qblk * 128 + wave * 16;

  // Q fragments (16x64) straight from global, A-layout
  v16bf aQ[2];
  {
    const u16* qrow = Qb + ((long)h * S_LEN + (q0 + lmod)) * 64;
    int c0 = lhalf ? 8 : 0;
    #pragma unroll
    for (int kc = 0; kc < 2; ++kc)
      aQ[kc] = ld_frag(qrow + kc * 32 + c0, qrow + kc * 32 + c0 + 16);
  }

#if HAVE_TDM
  const unsigned aKs = (unsigned)(size_t)(void*)&Ks[0];
#endif

  v8f o[4] = {};
  float m_i[8], l_i[8];
  #pragma unroll
  for (int r = 0; r < 8; ++r) { m_i[r] = -1e30f; l_i[r] = 0.f; }

  const int nTiles = (qblk + 1) * 4;       // causal: keys up to (qblk+1)*128
  for (int jt = 0; jt < nTiles; ++jt) {
    const int kbase = jt * 32;
    __syncthreads();
#if HAVE_TDM
    if (wave == 0)   // K tile (32x64 bf16) via Tensor Data Mover
      tdm_load_2d(aKs, Kb + (long)kbase * 64, 16, 32, 16);
#else
    {   // stage K tile (32 keys x 64) row-major
      int row = tid >> 3, cc = tid & 7;
      *(uint4*)&Ks[row * 64 + cc * 8] =
          *(const uint4*)(Kb + (long)(kbase + row) * 64 + cc * 8);
    }
#endif
    {   // stage V tile transposed: Vt[d][key]
      int row = tid >> 3, cc = tid & 7;
      U4U16 u;
      u.q = *(const uint4*)(Vb + (long)(kbase + row) * 64 + cc * 8);
      #pragma unroll
      for (int e = 0; e < 8; ++e)
        Vt[(cc * 8 + e) * 32 + row] = u.s[e];
    }
#if HAVE_TDM
    if (wave == 0) WAIT_TENSORCNT(0);
    asm volatile("" ::: "memory");
#endif
    __syncthreads();

    // S = Q K^T  (16 x 32)
    v8f sc[2] = {};
    #pragma unroll
    for (int nf = 0; nf < 2; ++nf) {
      int keyrow = nf * 16 + lmod;
      #pragma unroll
      for (int kc = 0; kc < 2; ++kc) {
        int ks = kc * 32 + (lhalf ? 8 : 0);
        v16bf bK = ld_frag(&Ks[keyrow * 64 + ks], &Ks[keyrow * 64 + ks + 16]);
        sc[nf] = wmma_bf16(aQ[kc], bK, sc[nf]);
      }
    }
    // scale 1/sqrt(64) + causal mask (== reference's -1e9 triu mask)
    #pragma unroll
    for (int nf = 0; nf < 2; ++nf) {
      int key = kbase + nf * 16 + lmod;
      #pragma unroll
      for (int r = 0; r < 8; ++r) {
        int q = q0 + r + 8 * lhalf;
        float v = sc[nf][r] * 0.125f;
        sc[nf][r] = (key > q) ? (v - 1e9f) : v;
      }
    }
    // online softmax: rows live across 16 lanes of one half -> shfl_xor 1..8
    float alpha[8];
    #pragma unroll
    for (int r = 0; r < 8; ++r) {
      float t = fmaxf(sc[0][r], sc[1][r]);
      #pragma unroll
      for (int off = 1; off < 16; off <<= 1)
        t = fmaxf(t, __shfl_xor(t, off, 32));
      float mnew = fmaxf(m_i[r], t);
      alpha[r] = __expf(m_i[r] - mnew);
      m_i[r]   = mnew;
      sc[0][r] = __expf(sc[0][r] - mnew);
      sc[1][r] = __expf(sc[1][r] - mnew);
      float su = sc[0][r] + sc[1][r];
      #pragma unroll
      for (int off = 1; off < 16; off <<= 1)
        su += __shfl_xor(su, off, 32);
      l_i[r] = l_i[r] * alpha[r] + su;
    }
    #pragma unroll
    for (int df = 0; df < 4; ++df)
      #pragma unroll
      for (int r = 0; r < 8; ++r)
        o[df][r] *= alpha[r];

    // P: C-layout -> per-wave LDS -> A-layout fragment (bf16)
    u16* myP = &Ps[wave][0];
    #pragma unroll
    for (int nf = 0; nf < 2; ++nf)
      #pragma unroll
      for (int r = 0; r < 8; ++r)
        myP[(r + 8 * lhalf) * 32 + nf * 16 + lmod] = f2bf(sc[nf][r]);
    asm volatile("s_wait_dscnt 0x0" ::: "memory");   // wave-internal LDS RAW
    int c0 = lhalf ? 8 : 0;
    v16bf aP = ld_frag(&myP[lmod * 32 + c0], &myP[lmod * 32 + c0 + 16]);

    // O += P * V   (P 16x32, V 32x64 -> 4 d-fragments)
    #pragma unroll
    for (int df = 0; df < 4; ++df) {
      int n  = df * 16 + lmod;
      int k0 = lhalf ? 8 : 0;
      v16bf bV = ld_frag(&Vt[n * 32 + k0], &Vt[n * 32 + k0 + 16]);
      o[df] = wmma_bf16(aP, bV, o[df]);
    }
  }

  // normalize and store bf16 [S][NH*64]
  #pragma unroll
  for (int df = 0; df < 4; ++df)
    #pragma unroll
    for (int r = 0; r < 8; ++r) {
      int row = q0 + r + 8 * lhalf;
      int col = h * 64 + df * 16 + lmod;
      Ob[(long)row * HID + col] = f2bf(o[df][r] / l_i[r]);
    }
}

// ---------------------------------------------------------------------------
extern "C" void kernel_launch(void* const* d_in, const int* in_sizes, int n_in,
                              void* d_out, int out_size, void* d_ws, size_t ws_size,
                              hipStream_t stream) {
  const float* x       = (const float*)d_in[0];
  const float* w_qkv   = (const float*)d_in[1];
  const float* w_dense = (const float*)d_in[2];
  // d_in[3] (mask) unused: causal mask computed inline. d_in[4] start_pos==0.

  char* ws = (char*)d_ws;
  size_t off = 0;
  auto alloc = [&](size_t bytes) {
    char* p = ws + off;
    off = (off + bytes + 255) & ~(size_t)255;
    return p;
  };
  u16*   xb   = (u16*)  alloc((size_t)S_LEN * HID * 2);                  // 18.6 MB
  u16*   wqb  = (u16*)  alloc((size_t)QKV_O * HID * 2);                  // 42.5 MB
  u16*   wdb  = (u16*)  alloc((size_t)HID * HID * 2);                    // 41.3 MB
  float* xqkv = (float*)alloc((size_t)S_LEN * QKV_O * sizeof(float));    // 38.3 MB
  u16*   Qb   = (u16*)  alloc((size_t)NHEAD * S_LEN * HDIM * 2);         // 18.6 MB
  u16*   Kb   = (u16*)  alloc((size_t)S_LEN * HDIM * 2);
  u16*   Vb   = (u16*)  alloc((size_t)S_LEN * HDIM * 2);
  u16*   Ob   = (u16*)  alloc((size_t)S_LEN * HID * 2);                  // 18.6 MB

  // 0) one-pass f32 -> bf16 conversion (weights then stay bf16-resident in L2)
  int nx = S_LEN * HID / 4, nq = QKV_O * HID / 4, nd = HID * HID / 4;
  cvt_bf16_kernel<<<(nx + 255) / 256, 256, 0, stream>>>((const float4*)x, (ushort4*)xb, nx);
  cvt_bf16_kernel<<<(nq + 255) / 256, 256, 0, stream>>>((const float4*)w_qkv, (ushort4*)wqb, nq);
  cvt_bf16_kernel<<<(nd + 255) / 256, 256, 0, stream>>>((const float4*)w_dense, (ushort4*)wdb, nd);

  // 1) QKV GEMM: xb[2048,4544] * wqb[4672,4544]^T -> xqkv f32
  gemm_kernel<<<dim3(QKV_O / 64, S_LEN / 128), 256, 0, stream>>>(
      xb, wqb, xqkv, S_LEN, QKV_O, HID);
  // 2) RoPE + head split + bf16
  rope_kernel<<<(S_LEN * 73 * 32) / 256, 256, 0, stream>>>(xqkv, Qb, Kb, Vb);
  // 3) causal flash attention
  flash_kernel<<<dim3(S_LEN / 128, NHEAD), 256, 0, stream>>>(Qb, Kb, Vb, Ob);
  // 4) dense GEMM: Ob[2048,4544] * wdb[4544,4544]^T -> out f32
  gemm_kernel<<<dim3(HID / 64, S_LEN / 128), 256, 0, stream>>>(
      Ob, wdb, (float*)d_out, S_LEN, HID, HID);
}
